// LSTMAutoEncoder_4501125726793
// MI455X (gfx1250) — matrix operand order repaired and checked by reading, not compiled, not run on hardware
//
#include <hip/hip_runtime.h>
#include <hip/hip_bf16.h>
#include <math.h>

// Problem constants (reference): B=128, T=128, F=128, D=512, gates=4D=2048
#define BB 128
#define TTS 128
#define FF 128
#define DD 512
#define GG 2048

#define HSTR 520   // padded LDS row stride for 512-wide h tiles (elements)
#define XSTR 136   // padded LDS row stride for 128-wide x tiles (elements)

typedef __bf16 bf16_t;
typedef __attribute__((ext_vector_type(16))) __bf16 v16bf;
typedef __attribute__((ext_vector_type(8)))  __bf16 v8bf;
typedef __attribute__((ext_vector_type(8)))  float  v8f;

__device__ __forceinline__ float sigmoidf_(float x) {
  return 1.0f / (1.0f + __expf(-x));
}

// A fragment 16x32 bf16: lane l: M=l&15, kb=(l>>4)*16;
// v[0..15] = K kb..kb+15 (contiguous)  -> two 16B loads.
__device__ __forceinline__ v16bf make_a(const bf16_t* p) {
  v8bf lo = *(const v8bf*)(p);
  v8bf hi = *(const v8bf*)(p + 8);
  v16bf a;
#pragma unroll
  for (int i = 0; i < 8; ++i) { a[i] = lo[i]; a[i + 8] = hi[i]; }
  return a;
}

// Accumulate 4 gate tiles at once: acc[g] += A(16xK) x B[g](Kx16).
// A is an LDS tile (row stride strideA); Bt is the (2048,K) bf16 row-major
// transposed weight; B-fragment for lane l: column n = g*512+jc, 16 contiguous
// K values starting at (l>=16 ? k+16 : k)  -> one 32B load.
__device__ __forceinline__ void gemm4(const bf16_t* __restrict__ shA, int strideA, int K,
                                      const bf16_t* __restrict__ Bt, int jc,
                                      int lane, v8f acc[4])
{
  const int kb8  = (lane >> 4) << 3;
  const int kb16 = (lane >> 4) << 4;
  const bf16_t* arow = shA + (lane & 15) * strideA;
#pragma unroll 2
  for (int k = 0; k < K; k += 32) {
    v16bf a = make_a(arow + k + kb16);
#pragma unroll
    for (int g = 0; g < 4; ++g) {
      v16bf b = *(const v16bf*)(Bt + (size_t)(g * DD + jc) * K + k + kb16);
      acc[g] = __builtin_amdgcn_wmma_f32_16x16x32_bf16(false, a, false, b,
                                                       (short)0, acc[g], false, false);
    }
  }
}

// LSTM gate math, fully in registers. acc[g][r] maps to the SAME (row, col)
// for every g (identical C/D layout), so this is elementwise per lane.
//   i=sig, f=sig, g=relu, o=sig; c2 = f*c + i*g; h2 = o*relu(c2)
__device__ __forceinline__ void lstm_gate(const v8f acc[4], const float bias[4],
                                          const v8f& c_in, v8f& c_out, v8f& h_out)
{
#pragma unroll
  for (int r = 0; r < 8; ++r) {
    const float i = sigmoidf_(acc[0][r] + bias[0]);
    const float f = sigmoidf_(acc[1][r] + bias[1]);
    const float g = fmaxf(acc[2][r] + bias[2], 0.0f);
    const float o = sigmoidf_(acc[3][r] + bias[3]);
    const float c2 = f * c_in[r] + i * g;
    c_out[r] = c2;
    h_out[r] = o * fmaxf(c2, 0.0f);
  }
}

// Scatter a 16x16 f32 tile (C/D layout) into an LDS h-buffer as bf16.
__device__ __forceinline__ void store_h(bf16_t* __restrict__ shHbuf, int jcol,
                                        int lane, const v8f& h)
{
  const int col = jcol + (lane & 15);
  const int rb  = (lane >> 4) << 3;
#pragma unroll
  for (int r = 0; r < 8; ++r)
    shHbuf[(rb + r) * HSTR + col] = (bf16_t)h[r];
}

// Cooperative load of a 16x128 f32 tile (row stride T*F) -> LDS bf16.
__device__ __forceinline__ void load_x(const float* __restrict__ src, int row0,
                                       bf16_t* __restrict__ shX, int tid)
{
#pragma unroll
  for (int it = 0; it < 2; ++it) {
    const int e = tid + it * 1024;
    const int r = e >> 7, f = e & 127;
    shX[r * XSTR + f] = (bf16_t)src[(size_t)(row0 + r) * (TTS * FF) + f];
  }
}

// ---------------------------------------------------------------------------
// Persistent LSTM autoencoder: grid = 8 blocks (16 batch rows each),
// block = 1024 threads = 32 waves; wave w owns hidden columns [16w, 16w+16).
// c-state never leaves registers; h round-trips through LDS only.
__global__ __launch_bounds__(1024)
void lstm_persist(const float* __restrict__ enc_in, const float* __restrict__ dec_in,
                  const bf16_t* __restrict__ encW0t, const bf16_t* __restrict__ encU0t,
                  const bf16_t* __restrict__ encWU1t,
                  const float* __restrict__ enc_b0, const float* __restrict__ enc_b1,
                  const bf16_t* __restrict__ decW0t, const bf16_t* __restrict__ decU0t,
                  const bf16_t* __restrict__ decW1t, const bf16_t* __restrict__ decU1t,
                  const float* __restrict__ dec_b0, const float* __restrict__ dec_b1,
                  const bf16_t* __restrict__ denseWt, const float* __restrict__ dense_b,
                  float* __restrict__ out)
{
  __shared__ bf16_t shH[3 * 16 * HSTR];   // three 16x512 h buffers (padded)
  __shared__ bf16_t shX[16 * XSTR];       // 16x128 input tile

  const int tid  = threadIdx.x;
  const int lane = tid & 31;
  const int wave = tid >> 5;
  const int row0 = blockIdx.x << 4;       // batch-row slice
  const int jcol = wave << 4;             // hidden-column base (0..511)
  const int jc   = jcol + (lane & 15);
  const int rb   = (lane >> 4) << 3;

  const int O0 = 0, O1 = 16 * HSTR, O2 = 32 * HSTR;

  // h0 = 0
  for (int e = tid; e < 16 * HSTR; e += 1024) shH[e] = (bf16_t)0.0f;

  // per-lane gate biases (constant over time)
  float eb0[4], eb1[4], db0[4], db1[4];
#pragma unroll
  for (int g = 0; g < 4; ++g) {
    eb0[g] = enc_b0[g * DD + jc];
    eb1[g] = enc_b1[g * DD + jc];
    db0[g] = dec_b0[g * DD + jc];
    db1[g] = dec_b1[g * DD + jc];
  }

  v8f c_carry = {};   // c0 = 0, lives in registers for all 256 steps

  // ---------------- encoder ----------------
  for (int t = 0; t < TTS; ++t) {
    load_x(enc_in + (size_t)t * FF, row0, shX, tid);
    __syncthreads();                       // x ready; prev-step h(O0) ready
    // layer 0: Z = x@W0 + h@U0 + b0
    v8f acc[4] = {};
    gemm4(shX, XSTR, FF, encW0t, jc, lane, acc);
    gemm4(shH + O0, HSTR, DD, encU0t, jc, lane, acc);
    v8f c1, h1;
    lstm_gate(acc, eb0, c_carry, c1, h1);
    store_h(shH + O1, jcol, lane, h1);
    __syncthreads();                       // h1 ready
    // layer 1: x == h1  =>  Z = h1@(W1+U1) + b1
    v8f acc2[4] = {};
    gemm4(shH + O1, HSTR, DD, encWU1t, jc, lane, acc2);
    v8f h2;
    lstm_gate(acc2, eb1, c1, c_carry, h2);
    store_h(shH + O0, jcol, lane, h2);     // new carry h (read after top barrier)
  }
  __syncthreads();

  // ---------------- decoder ----------------
  load_x(dec_in, row0, shX, tid);          // dec_in[:, 0, :]
  __syncthreads();
  int curO = O0, nxtO = O2;
  const float dbv = (wave < 8) ? dense_b[(wave << 4) + (lane & 15)] : 0.0f;

  for (int t = 0; t < TTS; ++t) {
    // layer 0: reads original carry (h,c); c2 discarded
    v8f acc[4] = {};
    gemm4(shX, XSTR, FF, decW0t, jc, lane, acc);
    gemm4(shH + curO, HSTR, DD, decU0t, jc, lane, acc);
    v8f cdead, h2a;
    lstm_gate(acc, db0, c_carry, cdead, h2a);
    store_h(shH + O1, jcol, lane, h2a);
    __syncthreads();                       // h2a ready
    // layer 1: also reads original carry (h,c); outputs become new carry
    v8f acc2[4] = {};
    gemm4(shH + O1, HSTR, DD, decW1t, jc, lane, acc2);
    gemm4(shH + curO, HSTR, DD, decU1t, jc, lane, acc2);
    v8f h2;
    lstm_gate(acc2, db1, c_carry, c_carry, h2);
    store_h(shH + nxtO, jcol, lane, h2);
    __syncthreads();                       // new h ready
    // dense readout (waves 0..7): out = h2 @ dense_W + dense_b,
    // written strided to d_out[:, T-1-t, :] and fed back as bf16 x.
    const int tt = TTS - 1 - t;
    if (wave < 8) {
      const int n = (wave << 4) + (lane & 15);
      const int kb8  = (lane >> 4) << 3;
      const int kb16 = (lane >> 4) << 4;
      const bf16_t* arow = shH + nxtO + (lane & 15) * HSTR;
      v8f acc3 = {};
#pragma unroll 2
      for (int k = 0; k < DD; k += 32) {
        v16bf a = make_a(arow + k + kb16);
        v16bf b = *(const v16bf*)(denseWt + (size_t)n * DD + k + kb16);
        acc3 = __builtin_amdgcn_wmma_f32_16x16x32_bf16(false, a, false, b,
                                                       (short)0, acc3, false, false);
      }
#pragma unroll
      for (int r = 0; r < 8; ++r) {
        const int row = rb + r;
        const float v = acc3[r] + dbv;
        out[(size_t)(row0 + row) * (TTS * FF) + (size_t)tt * FF + n] = v;
        shX[row * XSTR + n] = (bf16_t)v;
      }
    }
    __syncthreads();                       // feedback x ready; swap h buffers
    const int tmpO = curO; curO = nxtO; nxtO = tmpO;
  }
}

// ---------------------------------------------------------------------------
// Weight prep: f32 (K,N) [optionally summed with second (K,N)] -> bf16 (N,K).
__global__ __launch_bounds__(256)
void transpose_conv(const float* __restrict__ W, const float* __restrict__ W2,
                    bf16_t* __restrict__ Wt, int K, int N)
{
  __shared__ float tile[32][33];
  const int kb = blockIdx.y * 32;
  const int nb = blockIdx.x * 32;
  const int tx = threadIdx.x & 31;
  const int ty = threadIdx.x >> 5;
#pragma unroll
  for (int r = ty; r < 32; r += 8) {
    float v = W[(size_t)(kb + r) * N + nb + tx];
    if (W2) v += W2[(size_t)(kb + r) * N + nb + tx];
    tile[r][tx] = v;
  }
  __syncthreads();
#pragma unroll
  for (int r = ty; r < 32; r += 8) {
    Wt[(size_t)(nb + r) * K + kb + tx] = (bf16_t)tile[tx][r];
  }
}

// ---------------------------------------------------------------------------
static inline size_t alignup(size_t x) { return (x + 255) & ~(size_t)255; }

extern "C" void kernel_launch(void* const* d_in, const int* in_sizes, int n_in,
                              void* d_out, int out_size, void* d_ws, size_t ws_size,
                              hipStream_t stream) {
  const float* enc_in  = (const float*)d_in[0];
  const float* dec_in  = (const float*)d_in[1];
  const float* enc_W0  = (const float*)d_in[2];
  const float* enc_U0  = (const float*)d_in[3];
  const float* enc_b0  = (const float*)d_in[4];
  const float* enc_W1  = (const float*)d_in[5];
  const float* enc_U1  = (const float*)d_in[6];
  const float* enc_b1  = (const float*)d_in[7];
  const float* dec_W0  = (const float*)d_in[8];
  const float* dec_U0  = (const float*)d_in[9];
  const float* dec_b0  = (const float*)d_in[10];
  const float* dec_W1  = (const float*)d_in[11];
  const float* dec_U1  = (const float*)d_in[12];
  const float* dec_b1  = (const float*)d_in[13];
  const float* dense_W = (const float*)d_in[14];
  const float* dense_b = (const float*)d_in[15];
  float* out = (float*)d_out;
  (void)in_sizes; (void)n_in; (void)out_size; (void)ws_size;

  // workspace: bf16 transposed weights only (~11 MB, L2-resident)
  char* ws = (char*)d_ws;
  size_t off = 0;
  auto take = [&](size_t bytes) { char* p = ws + off; off = alignup(off + bytes); return p; };

  bf16_t* encW0t  = (bf16_t*)take((size_t)GG * FF * 2);   // (2048,128)
  bf16_t* encU0t  = (bf16_t*)take((size_t)GG * DD * 2);   // (2048,512)
  bf16_t* encWU1t = (bf16_t*)take((size_t)GG * DD * 2);   // (W1+U1)^T
  bf16_t* decW0t  = (bf16_t*)take((size_t)GG * FF * 2);
  bf16_t* decU0t  = (bf16_t*)take((size_t)GG * DD * 2);
  bf16_t* decW1t  = (bf16_t*)take((size_t)GG * DD * 2);
  bf16_t* decU1t  = (bf16_t*)take((size_t)GG * DD * 2);
  bf16_t* denseWt = (bf16_t*)take((size_t)FF * DD * 2);   // (128,512)

  const dim3 blk(256);
  transpose_conv<<<dim3(GG/32, FF/32), blk, 0, stream>>>(enc_W0, nullptr, encW0t, FF, GG);
  transpose_conv<<<dim3(GG/32, DD/32), blk, 0, stream>>>(enc_U0, nullptr, encU0t, DD, GG);
  transpose_conv<<<dim3(GG/32, DD/32), blk, 0, stream>>>(enc_W1, enc_U1,  encWU1t, DD, GG);
  transpose_conv<<<dim3(GG/32, FF/32), blk, 0, stream>>>(dec_W0, nullptr, decW0t, FF, GG);
  transpose_conv<<<dim3(GG/32, DD/32), blk, 0, stream>>>(dec_U0, nullptr, decU0t, DD, GG);
  transpose_conv<<<dim3(GG/32, DD/32), blk, 0, stream>>>(dec_W1, nullptr, decW1t, DD, GG);
  transpose_conv<<<dim3(GG/32, DD/32), blk, 0, stream>>>(dec_U1, nullptr, decU1t, DD, GG);
  transpose_conv<<<dim3(FF/32, DD/32), blk, 0, stream>>>(dense_W, nullptr, denseWt, DD, FF);

  lstm_persist<<<dim3(BB / 16), dim3(1024), 0, stream>>>(
      enc_in, dec_in,
      encW0t, encU0t, encWU1t, enc_b0, enc_b1,
      decW0t, decU0t, decW1t, decU1t, dec_b0, dec_b1,
      denseWt, dense_b, out);
}
